// ShapeSensitiveLoss_80985903334014
// MI455X (gfx1250) — matrix operand order repaired and checked by reading, not compiled, not run on hardware
//
#include <hip/hip_runtime.h>
#include <hip/hip_bf16.h>
#include <math.h>

// ---------------------------------------------------------------------------
// Types for CDNA5 WMMA (wave32, v_wmma_f32_16x16x32_f16)
// ---------------------------------------------------------------------------
typedef __attribute__((ext_vector_type(16))) _Float16 v16h;
typedef __attribute__((ext_vector_type(8)))  _Float16 v8h;
typedef __attribute__((ext_vector_type(8)))  float    v8f;

union HFrag { v16h v; v8h h[2]; };

#define BM 128
#define BN 128
#define BK 32
#define LDSK 48   // halves per LDS row: 96B => every 8-half (16B) access aligned

__device__ __forceinline__ float gelu_exact(float x) {
  return 0.5f * x * (1.0f + erff(x * 0.70710678118654752440f));
}

// Load 16 consecutive K-elements of one row into registers (f32).
// Fast path unguarded; slow path (tile edges only) guarded with zero fill.
__device__ __forceinline__ void load_row16(
    const float* __restrict__ G, long ld, int gr, int RMax, int k0, int K,
    float* __restrict__ r) {
  if (gr < RMax && (k0 + 16) <= K) {
    const float* src = G + (long)gr * ld + k0;
#pragma unroll
    for (int i = 0; i < 16; ++i) r[i] = src[i];
  } else {
#pragma unroll
    for (int i = 0; i < 16; ++i) {
      const int gk = k0 + i;
      r[i] = (gr < RMax && gk < K) ? G[(long)gr * ld + gk] : 0.0f;
    }
  }
}

// Convert 16 f32 -> f16 and store as two 16B ds_store_b128.
__device__ __forceinline__ void store_row16(
    _Float16* __restrict__ S, int srow, int kb, const float* __restrict__ r) {
  v8h h0, h1;
#pragma unroll
  for (int i = 0; i < 8; ++i) h0[i] = (_Float16)r[i];
#pragma unroll
  for (int i = 0; i < 8; ++i) h1[i] = (_Float16)r[8 + i];
  *(v8h*)&S[srow * LDSK + kb]     = h0;
  *(v8h*)&S[srow * LDSK + kb + 8] = h1;
}

// ---------------------------------------------------------------------------
// Generic batched WMMA GEMM:  C[z] = act(scale * A[z] @ B[z]^T + bias) + resid
// A: M x K row-major (lda).  B: N x K row-major (ldb)  [output column
// contiguous in K -- weights pre-transposed on device once per layer].
// z = grid.z decomposes as (b = z/H, h = z%H) with per-b / per-h strides.
// 128x128 block, 8 waves (2x4), each wave 64x32 = 8 WMMAs per K-chunk.
// LDS double-buffered: global loads for chunk k+1 are issued into registers
// while chunk k is computed, then converted+stored; one barrier/iteration.
// ---------------------------------------------------------------------------
__global__ __launch_bounds__(256) void wmma_gemm_kernel(
    const float* __restrict__ A, long lda, long sAb, long sAh,
    const float* __restrict__ B, long ldb, long sBb, long sBh,
    float* __restrict__ C, long ldc, long sCb, long sCh,
    const float* __restrict__ bias, const float* __restrict__ resid,
    int M, int N, int K, int H, float scale, int act) {
  const int z = blockIdx.z;
  const int zb = z / H, zh = z % H;
  const float* Ag = A + (long)zb * sAb + (long)zh * sAh;
  const float* Bg = B + (long)zb * sBb + (long)zh * sBh;
  float*       Cg = C + (long)zb * sCb + (long)zh * sCh;
  const float* Rg = resid ? (resid + (long)zb * sCb + (long)zh * sCh) : nullptr;

  __shared__ __align__(16) _Float16 Ash[2][BM * LDSK];
  __shared__ __align__(16) _Float16 Bsh[2][BN * LDSK];

  const int row0 = blockIdx.x * BM;
  const int col0 = blockIdx.y * BN;
  const int tid  = threadIdx.x;
  const int wave = tid >> 5;
  const int lane = tid & 31;
  const int wm   = (wave >> 2) * 64;   // 2 wave-rows of 64
  const int wn   = (wave & 3) * 32;    // 4 wave-cols of 32
  const int hlf  = lane >> 4;          // 0: lanes 0-15, 1: lanes 16-31
  const int l16  = lane & 15;

  // staging assignment (identical pattern for A and B tiles: 128 rows x 32 K)
  const int srow = tid >> 1;           // 0..127
  const int skb  = (tid & 1) * 16;     // 0 or 16
  const int gmA  = row0 + srow;
  const int gnB  = col0 + srow;

  v8f acc[4][2];
#pragma unroll
  for (int i = 0; i < 4; ++i)
#pragma unroll
    for (int j = 0; j < 2; ++j)
#pragma unroll
      for (int e = 0; e < 8; ++e) acc[i][j][e] = 0.0f;

  const int nk = (K + BK - 1) / BK;
  float ra[16], rb[16];

  // prologue: stage chunk 0 into buffer 0
  load_row16(Ag, lda, gmA, M, skb, K, ra);
  load_row16(Bg, ldb, gnB, N, skb, K, rb);
  store_row16(Ash[0], srow, skb, ra);
  store_row16(Bsh[0], srow, skb, rb);
  __syncthreads();

  for (int kc = 0; kc < nk; ++kc) {
    const int cur = kc & 1;
    // issue next chunk's global loads (latency hidden behind WMMAs below)
    if (kc + 1 < nk) {
      const int k0n = (kc + 1) * BK + skb;
      if (kc + 2 < nk && gmA < M)
        __builtin_prefetch(Ag + (long)gmA * lda + (kc + 2) * BK + skb, 0, 3);
      load_row16(Ag, lda, gmA, M, k0n, K, ra);
      load_row16(Bg, ldb, gnB, N, k0n, K, rb);
    }

    // fragments per documented 16-bit WMMA layouts
    const _Float16* As = Ash[cur];
    const _Float16* Bs = Bsh[cur];
    HFrag af[4], bf[2];
#pragma unroll
    for (int ti = 0; ti < 4; ++ti) {
      const int m  = wm + ti * 16 + l16;
      const int ko = hlf * 8;   // lanes 0-15: K0-7/K16-23; lanes 16-31: K8-15/K24-31
      af[ti].h[0] = *(const v8h*)&As[m * LDSK + ko];
      af[ti].h[1] = *(const v8h*)&As[m * LDSK + ko + 16];
    }
#pragma unroll
    for (int tj = 0; tj < 2; ++tj) {
      const int n  = wn + tj * 16 + l16;
      const int kb = hlf * 8;   // lanes 0-15: K0-7/K16-23; lanes 16-31: K8-15/K24-31
      bf[tj].h[0] = *(const v8h*)&Bs[n * LDSK + kb];
      bf[tj].h[1] = *(const v8h*)&Bs[n * LDSK + kb + 16];
    }
#pragma unroll
    for (int ti = 0; ti < 4; ++ti)
#pragma unroll
      for (int tj = 0; tj < 2; ++tj)
        acc[ti][tj] = __builtin_amdgcn_wmma_f32_16x16x32_f16(
            false, af[ti].v, false, bf[tj].v, (short)0, acc[ti][tj], false, false);

    // stage next chunk into the other buffer
    if (kc + 1 < nk) {
      store_row16(Ash[cur ^ 1], srow, skb, ra);
      store_row16(Bsh[cur ^ 1], srow, skb, rb);
      __syncthreads();
    }
  }

  // ---- epilogue: scale, bias, activation, residual ----
#pragma unroll
  for (int ti = 0; ti < 4; ++ti)
#pragma unroll
    for (int tj = 0; tj < 2; ++tj)
#pragma unroll
      for (int r = 0; r < 8; ++r) {
        const int gm = row0 + wm + ti * 16 + r + hlf * 8;
        const int gn = col0 + wn + tj * 16 + l16;
        if (gm < M && gn < N) {
          float v = acc[ti][tj][r] * scale;
          if (bias) v += bias[gn];
          if (act == 1) v = gelu_exact(v);
          if (Rg) v += Rg[(long)gm * ldc + gn];
          Cg[(long)gm * ldc + gn] = v;
        }
      }
}

// ---------------------------------------------------------------------------
// Batched tiled f32 transpose: out[c][r] = in[r][c]. 32x32 LDS tiles,
// coalesced both directions. Pre-transposes weights (once per layer) and V
// so GEMM B-staging is always contiguous in K.
// ---------------------------------------------------------------------------
__global__ __launch_bounds__(256) void transpose_f32(
    const float* __restrict__ in, long ldin, long sib, long sih,
    float* __restrict__ out, long ldo, long sob, long soh,
    int R, int C, int H) {
  const int z = blockIdx.z;
  const int zb = z / H, zh = z % H;
  const float* I = in + (long)zb * sib + (long)zh * sih;
  float* O = out + (long)zb * sob + (long)zh * soh;
  __shared__ float tile[32][33];
  const int tx = threadIdx.x & 31;
  const int ty = threadIdx.x >> 5;           // 8 rows of threads
  const int c0 = blockIdx.x * 32, r0 = blockIdx.y * 32;
#pragma unroll
  for (int i = 0; i < 32; i += 8) {
    const int r = r0 + ty + i, c = c0 + tx;
    tile[ty + i][tx] = (r < R && c < C) ? I[(long)r * ldin + c] : 0.0f;
  }
  __syncthreads();
#pragma unroll
  for (int i = 0; i < 32; i += 8) {
    const int c = c0 + ty + i, r = r0 + tx;
    if (c < C && r < R) O[(long)c * ldo + r] = tile[tx][ty + i];
  }
}

// ---------------------------------------------------------------------------
// LayerNorm over D=768, one row per block (strides parameterized).
// ---------------------------------------------------------------------------
__global__ __launch_bounds__(256) void layernorm_rows(
    const float* __restrict__ X, long xstride,
    float* __restrict__ Y, long ystride,
    const float* __restrict__ g, const float* __restrict__ b, int D) {
  const long row = blockIdx.x;
  const float* x = X + row * xstride;
  float* y = Y + row * ystride;
  __shared__ float red[256];
  const int tid = threadIdx.x;

  float s = 0.0f;
  for (int i = tid; i < D; i += 256) s += x[i];
  red[tid] = s; __syncthreads();
  for (int st = 128; st > 0; st >>= 1) { if (tid < st) red[tid] += red[tid + st]; __syncthreads(); }
  const float mu = red[0] / (float)D;
  __syncthreads();

  float v = 0.0f;
  for (int i = tid; i < D; i += 256) { float d = x[i] - mu; v += d * d; }
  red[tid] = v; __syncthreads();
  for (int st = 128; st > 0; st >>= 1) { if (tid < st) red[tid] += red[tid + st]; __syncthreads(); }
  const float inv = rsqrtf(red[0] / (float)D + 1e-6f);
  __syncthreads();

  for (int i = tid; i < D; i += 256) y[i] = (x[i] - mu) * inv * g[i] + b[i];
}

// ---------------------------------------------------------------------------
// In-place softmax over rows of length n (attention rows: 48*577 x 577).
// ---------------------------------------------------------------------------
__global__ __launch_bounds__(256) void softmax_rows(float* __restrict__ S, long rstride, int n) {
  float* s = S + (long)blockIdx.x * rstride;
  __shared__ float red[256];
  const int tid = threadIdx.x;

  float m = -3.4e38f;
  for (int i = tid; i < n; i += 256) m = fmaxf(m, s[i]);
  red[tid] = m; __syncthreads();
  for (int st = 128; st > 0; st >>= 1) { if (tid < st) red[tid] = fmaxf(red[tid], red[tid + st]); __syncthreads(); }
  m = red[0]; __syncthreads();

  float sum = 0.0f;
  for (int i = tid; i < n; i += 256) { float e = expf(s[i] - m); s[i] = e; sum += e; }
  red[tid] = sum; __syncthreads();
  for (int st = 128; st > 0; st >>= 1) { if (tid < st) red[tid] += red[tid + st]; __syncthreads(); }
  const float inv = 1.0f / red[0];
  __syncthreads();
  for (int i = tid; i < n; i += 256) s[i] *= inv;
}

// ---------------------------------------------------------------------------
// Masks: softmax(logits)[:,1] > 0.5  <=>  l1 > l0.  maps[4][384*384] =
// (pred b0, pred b1, true b0, true b1).
// ---------------------------------------------------------------------------
__global__ void build_masks(const float* __restrict__ logits, const int* __restrict__ tr,
                            float* __restrict__ masks) {
  const long HW = 384L * 384L;
  long idx = (long)blockIdx.x * blockDim.x + threadIdx.x;
  if (idx >= 2 * HW) return;
  const long b = idx / HW, p = idx % HW;
  const float l0 = logits[b * 2 * HW + p];
  const float l1 = logits[b * 2 * HW + HW + p];
  masks[b * HW + p]       = (l1 > l0) ? 1.0f : 0.0f;
  masks[(2 + b) * HW + p] = tr[b * HW + p] ? 1.0f : 0.0f;
}

// ---------------------------------------------------------------------------
// EDT row pass: squared distance along W to nearest 1 (g2a) / nearest 0 (g2b),
// capped at BIG=1e4 to match the reference.
// ---------------------------------------------------------------------------
__global__ void edt_rows(const float* __restrict__ masks,
                         float* __restrict__ g2a, float* __restrict__ g2b) {
  const int t = blockIdx.x * blockDim.x + threadIdx.x;
  if (t >= 4 * 384) return;
  const float* m = masks + (long)t * 384;
  float* ga = g2a + (long)t * 384;
  float* gb = g2b + (long)t * 384;
  const float BIG = 1e4f;
  float da = BIG, db = BIG;
  for (int x = 0; x < 384; ++x) {
    da = (m[x] != 0.0f) ? 0.0f : fminf(da + 1.0f, BIG);
    db = (m[x] == 0.0f) ? 0.0f : fminf(db + 1.0f, BIG);
    ga[x] = da; gb[x] = db;
  }
  da = ga[383]; db = gb[383];
  for (int x = 382; x >= 0; --x) {
    da = fminf(da + 1.0f, ga[x]); ga[x] = da;
    db = fminf(db + 1.0f, gb[x]); gb[x] = db;
  }
  for (int x = 0; x < 384; ++x) { ga[x] *= ga[x]; gb[x] *= gb[x]; }
}

// ---------------------------------------------------------------------------
// EDT column pass + signed distance.
// ---------------------------------------------------------------------------
__global__ void edt_cols_sdm(const float* __restrict__ g2a, const float* __restrict__ g2b,
                             float* __restrict__ sdm) {
  const int x = blockIdx.x * 64 + threadIdx.x;
  const int y = blockIdx.y;
  const int i = blockIdx.z;
  if (x >= 384) return;
  const float* ga = g2a + ((long)i * 384) * 384 + x;
  const float* gb = g2b + ((long)i * 384) * 384 + x;
  float ma = 3.4e38f, mb = 3.4e38f;
  for (int yy = 0; yy < 384; ++yy) {
    const float dy = (float)(y - yy);
    const float d2 = dy * dy;
    ma = fminf(ma, ga[(long)yy * 384] + d2);
    mb = fminf(mb, gb[(long)yy * 384] + d2);
  }
  sdm[((long)i * 384 + y) * 384 + x] = sqrtf(ma) - sqrtf(mb);
}

// ---------------------------------------------------------------------------
// Patchify: P[(b*576+ph*24+pw)*768 + c*256+iy*16+ix] = sdm[b, ph*16+iy, pw*16+ix]
// ---------------------------------------------------------------------------
__global__ void patchify(const float* __restrict__ sdm, float* __restrict__ P) {
  long idx = (long)blockIdx.x * blockDim.x + threadIdx.x;
  if (idx >= 2304L * 768L) return;
  const long rowi = idx / 768;
  const int f  = (int)(idx % 768);
  const int b  = (int)(rowi / 576);
  const int pi = (int)(rowi % 576);
  const int ph = pi / 24, pw = pi % 24;
  const int iy = (f >> 4) & 15, ix = f & 15;   // channel (f>>8) is a replica
  P[idx] = sdm[((long)b * 384 + ph * 16 + iy) * 384 + pw * 16 + ix];
}

// ---------------------------------------------------------------------------
// X[b,0,:] = cls; then X[b,t,:] += pos[t,:]  (X is [4,577,768])
// ---------------------------------------------------------------------------
__global__ void add_cls_pos(float* __restrict__ X, const float* __restrict__ cls,
                            const float* __restrict__ pos) {
  long idx = (long)blockIdx.x * blockDim.x + threadIdx.x;
  if (idx >= 4L * 577L * 768L) return;
  const int d = (int)(idx % 768);
  const int t = (int)((idx / 768) % 577);
  float v = (t == 0) ? cls[d] : X[idx];
  X[idx] = v + pos[(long)t * 768 + d];
}

// ---------------------------------------------------------------------------
// loss = 1 - mean_b cos(F[b], F[2+b]), eps-clamped norms. F is [4][768].
// ---------------------------------------------------------------------------
__global__ __launch_bounds__(256) void cosine_loss(const float* __restrict__ F,
                                                   float* __restrict__ out) {
  __shared__ float sd[256], sa[256], sb[256];
  const int tid = threadIdx.x;
  float closs = 0.0f;
  for (int p = 0; p < 2; ++p) {
    float d = 0.0f, a = 0.0f, b = 0.0f;
    for (int i = tid; i < 768; i += 256) {
      const float x = F[p * 768 + i], y = F[(2 + p) * 768 + i];
      d += x * y; a += x * x; b += y * y;
    }
    sd[tid] = d; sa[tid] = a; sb[tid] = b; __syncthreads();
    for (int st = 128; st > 0; st >>= 1) {
      if (tid < st) { sd[tid] += sd[tid + st]; sa[tid] += sa[tid + st]; sb[tid] += sb[tid + st]; }
      __syncthreads();
    }
    if (tid == 0) {
      const float na = fmaxf(sqrtf(sa[0]), 1e-8f);
      const float nb = fmaxf(sqrtf(sb[0]), 1e-8f);
      closs += sd[0] / (na * nb);
    }
    __syncthreads();
  }
  if (tid == 0) out[0] = 1.0f - 0.5f * closs;
}

// ---------------------------------------------------------------------------
// Host orchestration
// ---------------------------------------------------------------------------
extern "C" void kernel_launch(void* const* d_in, const int* in_sizes, int n_in,
                              void* d_out, int out_size, void* d_ws, size_t ws_size,
                              hipStream_t stream) {
  (void)in_sizes; (void)n_in; (void)out_size; (void)ws_size;
  const float* pred_logits = (const float*)d_in[0];
  const int*   truev       = (const int*)d_in[1];
  const float* patch_w     = (const float*)d_in[2];
  const float* patch_b     = (const float*)d_in[3];
  const float* cls_tok     = (const float*)d_in[4];
  const float* pos_emb     = (const float*)d_in[5];
  const float* ln1_g = (const float*)d_in[6],  *ln1_b = (const float*)d_in[7];
  const float* qkv_w = (const float*)d_in[8],  *qkv_b = (const float*)d_in[9];
  const float* proj_w = (const float*)d_in[10], *proj_b = (const float*)d_in[11];
  const float* ln2_g = (const float*)d_in[12], *ln2_b = (const float*)d_in[13];
  const float* fc1_w = (const float*)d_in[14], *fc1_b = (const float*)d_in[15];
  const float* fc2_w = (const float*)d_in[16], *fc2_b = (const float*)d_in[17];
  const float* norm_g = (const float*)d_in[18], *norm_b = (const float*)d_in[19];
  float* out = (float*)d_out;

  char* wp = (char*)d_ws;
  auto carve = [&](long nfloats) -> float* {
    float* p = (float*)wp;
    wp += ((nfloats * 4 + 255) / 256) * 256;
    return p;
  };
  const long HW = 384L * 384L;
  float* masks = carve(4 * HW);
  float* g2a   = carve(4 * HW);
  float* g2b   = carve(4 * HW);
  float* sdm   = carve(4 * HW);
  float* Pmat  = carve(2304L * 768);
  float* X     = carve(4L * 577 * 768);
  float* Hb    = carve(4L * 577 * 768);
  float* QKV   = carve(4L * 577 * 2304);
  float* Sc    = carve(48L * 577 * 577);
  float* Ob    = carve(4L * 577 * 768);
  float* M1    = carve(4L * 577 * 3072);
  float* Wt    = carve(768L * 3072);          // reusable transposed-weight buffer
  float* Vt    = carve(4L * 12 * 64 * 577);   // V^T per (b,h): [d][tokens]
  float* Fcls  = carve(4L * 768);

  auto gemm = [&](const float* A, long lda, long sAb, long sAh,
                  const float* B, long ldb, long sBb, long sBh,
                  float* C, long ldc, long sCb, long sCh,
                  const float* bias, const float* resid,
                  int M, int N, int K, int Z, int H, float scale, int act) {
    dim3 g((unsigned)((M + BM - 1) / BM), (unsigned)((N + BN - 1) / BN), (unsigned)Z);
    wmma_gemm_kernel<<<g, 256, 0, stream>>>(A, lda, sAb, sAh, B, ldb, sBb, sBh,
                                            C, ldc, sCb, sCh, bias, resid,
                                            M, N, K, H, scale, act);
  };
  auto transpose = [&](const float* in, long ldin, long sib, long sih,
                       float* outp, long ldo, long sob, long soh,
                       int R, int C, int Z, int H) {
    dim3 g((unsigned)((C + 31) / 32), (unsigned)((R + 31) / 32), (unsigned)Z);
    transpose_f32<<<g, 256, 0, stream>>>(in, ldin, sib, sih, outp, ldo, sob, soh, R, C, H);
  };

  // ---- Stage 1: masks -> EDT -> signed distance maps [4][384][384] ----
  build_masks<<<(unsigned)((2 * HW + 255) / 256), 256, 0, stream>>>(pred_logits, truev, masks);
  edt_rows<<<(4 * 384 + 255) / 256, 256, 0, stream>>>(masks, g2a, g2b);
  edt_cols_sdm<<<dim3(6, 384, 4), 64, 0, stream>>>(g2a, g2b, sdm);

  // ---- Stage 2: patch embed + cls + pos ----
  patchify<<<(unsigned)((2304L * 768 + 255) / 256), 256, 0, stream>>>(sdm, Pmat);
  transpose(patch_w, 768, 0, 0, Wt, 768, 0, 0, 768, 768, 1, 1);
  gemm(Pmat, 768, 576L * 768, 0,
       Wt, 768, 0, 0,
       X + 768, 768, 577L * 768, 0,
       patch_b, nullptr, 576, 768, 768, /*Z=*/4, /*H=*/1, 1.0f, 0);
  add_cls_pos<<<(unsigned)((4L * 577 * 768 + 255) / 256), 256, 0, stream>>>(X, cls_tok, pos_emb);

  // ---- Stage 3: 12 transformer blocks ----
  const int ROWS = 4 * 577;  // 2308 flattened tokens
  for (int L = 0; L < 12; ++L) {
    // LN1
    layernorm_rows<<<ROWS, 256, 0, stream>>>(X, 768, Hb, 768, ln1_g + L * 768, ln1_b + L * 768, 768);
    // QKV = Hb @ qkv_w[L] + qkv_b[L]   (weight pre-transposed to [2304][768])
    transpose(qkv_w + (long)L * 768 * 2304, 2304, 0, 0, Wt, 768, 0, 0, 768, 2304, 1, 1);
    gemm(Hb, 768, 0, 0,
         Wt, 768, 0, 0,
         QKV, 2304, 0, 0,
         qkv_b + (long)L * 2304, nullptr, ROWS, 2304, 768, 1, 1, 1.0f, 0);
    // scores[b,h] = 0.125 * Q @ K^T   (K rows already [token][d] = [N][K])
    gemm(QKV, 2304, 577L * 2304, 64,
         QKV + 768, 2304, 577L * 2304, 64,
         Sc, 577, 12L * 577 * 577, 577L * 577,
         nullptr, nullptr, 577, 577, 64, 48, 12, 0.125f, 0);
    softmax_rows<<<48 * 577, 256, 0, stream>>>(Sc, 577, 577);
    // V^T per head: Vt[b,h][d][t]
    transpose(QKV + 1536, 2304, 577L * 2304, 64,
              Vt, 577, 12L * 64 * 577, 64L * 577, 577, 64, 48, 12);
    // Ob[b, :, h*64:(h+1)*64] = P @ V
    gemm(Sc, 577, 12L * 577 * 577, 577L * 577,
         Vt, 577, 12L * 64 * 577, 64L * 577,
         Ob, 768, 577L * 768, 64,
         nullptr, nullptr, 577, 64, 577, 48, 12, 1.0f, 0);
    // X += Ob @ proj_w[L] + proj_b[L]
    transpose(proj_w + (long)L * 768 * 768, 768, 0, 0, Wt, 768, 0, 0, 768, 768, 1, 1);
    gemm(Ob, 768, 0, 0,
         Wt, 768, 0, 0,
         X, 768, 0, 0,
         proj_b + (long)L * 768, X, ROWS, 768, 768, 1, 1, 1.0f, 0);
    // LN2
    layernorm_rows<<<ROWS, 256, 0, stream>>>(X, 768, Hb, 768, ln2_g + L * 768, ln2_b + L * 768, 768);
    // M1 = gelu(Hb @ fc1_w[L] + fc1_b[L])
    transpose(fc1_w + (long)L * 768 * 3072, 3072, 0, 0, Wt, 768, 0, 0, 768, 3072, 1, 1);
    gemm(Hb, 768, 0, 0,
         Wt, 768, 0, 0,
         M1, 3072, 0, 0,
         fc1_b + (long)L * 3072, nullptr, ROWS, 3072, 768, 1, 1, 1.0f, /*gelu=*/1);
    // X += M1 @ fc2_w[L] + fc2_b[L]
    transpose(fc2_w + (long)L * 3072 * 768, 768, 0, 0, Wt, 3072, 0, 0, 3072, 768, 1, 1);
    gemm(M1, 3072, 0, 0,
         Wt, 3072, 0, 0,
         X, 768, 0, 0,
         fc2_b + (long)L * 768, X, ROWS, 768, 3072, 1, 1, 1.0f, 0);
  }

  // ---- Stage 4: final LN on the 4 CLS rows, cosine loss ----
  layernorm_rows<<<4, 256, 0, stream>>>(X, 577L * 768, Fcls, 768, norm_g, norm_b, 768);
  cosine_loss<<<1, 256, 0, stream>>>(Fcls, out);
}
